// SwinEncoderDecoderTransformer3D_5377299054936
// MI455X (gfx1250) — matrix operand order repaired and checked by reading, not compiled, not run on hardware
//
#include <hip/hip_runtime.h>
#include <hip/hip_bf16.h>
#include <math.h>

// ---------------------------------------------------------------------------
// Swin-3D block for MI455X (gfx1250, wave32, WMMA).
// All GEMMs run on v_wmma_f32_16x16x32_bf16 (bf16 in, f32 accumulate).
// Attention stages K/V through LDS with GLOBAL_LOAD_ASYNC_TO_LDS_B128.
// ---------------------------------------------------------------------------

typedef __bf16 bf16_t;
typedef __attribute__((ext_vector_type(16))) __bf16 v16bf;
typedef __attribute__((ext_vector_type(8)))  float  v8f;

// Problem constants
#define BATCH   4
#define DD      8
#define HH      56
#define WW_     56
#define CC      128
#define HEADS_  4
#define HD      32
#define NWIN    1024        // 4 * (8/2) * (56/7) * (56/7)
#define NPAD    128         // window tokens padded 98 -> 128
#define NTOK    98
#define TOKENS  100352      // BATCH*DD*HH*WW_
#define HIDDEN  512
#define MLP_CHUNK_ROWS 12544  // TOKENS / 8

__device__ __forceinline__ v8f zero8() {
  v8f z = {0.f, 0.f, 0.f, 0.f, 0.f, 0.f, 0.f, 0.f};
  return z;
}

// A-operand: lane holds row (lane&15); lanes<16 carry K {0..7,16..23},
// lanes>=16 carry K {8..15,24..31}  -> two 16B loads at +0 and +16 elems.
__device__ __forceinline__ v16bf ld_a16(const bf16_t* p) {
  union { uint4 q[2]; v16bf v; } u;
  u.q[0] = *reinterpret_cast<const uint4*>(p);
  u.q[1] = *reinterpret_cast<const uint4*>(p + 16);
  return u.v;
}

// B-operand: lane holds column (lane&15); lanes<16 carry K {0..7,16..23},
// lanes>=16 carry K {8..15,24..31} -> two 16B loads at +0 and +16 elems.
__device__ __forceinline__ v16bf ld_b16(const bf16_t* p) {
  union { uint4 q[2]; v16bf v; } u;
  u.q[0] = *reinterpret_cast<const uint4*>(p);
  u.q[1] = *reinterpret_cast<const uint4*>(p + 16);
  return u.v;
}

// Register-blocked 16 x (16*NT) tile block: A fragment loaded once per K-step
// and reused across NT WMMAs (4x less A traffic).
template <int NT>
__device__ __forceinline__ void wmma_rowblk(const bf16_t* __restrict__ A, int lda,
                                            const bf16_t* __restrict__ Bt, int ldb,
                                            int row0, int col0, int K, int lane,
                                            v8f acc[NT]) {
  const bf16_t* ap = A  + (size_t)(row0 + (lane & 15)) * lda + ((lane & 16) ? 8  : 0);
  const bf16_t* bp = Bt + (size_t)(col0 + (lane & 15)) * ldb + ((lane & 16) ? 8 : 0);
  for (int k = 0; k < K; k += 32) {
    v16bf a = ld_a16(ap + k);
#pragma unroll
    for (int t = 0; t < NT; ++t) {
      v16bf b = ld_b16(bp + (size_t)t * 16 * ldb + k);
      acc[t] = __builtin_amdgcn_wmma_f32_16x16x32_bf16(false, a, false, b, (short)0,
                                                       acc[t], false, false);
    }
  }
}

__device__ __forceinline__ float wave_sum32(float v) {
  for (int m = 16; m; m >>= 1) v += __shfl_xor(v, m, 32);
  return v;
}

// LayerNorm one 128-float row -> bf16, one wave per row (4 ch / lane).
__device__ __forceinline__ void ln_row(const float* __restrict__ src,
                                       const float* __restrict__ g,
                                       const float* __restrict__ b,
                                       bf16_t* __restrict__ dst, int lane) {
  float4 v  = reinterpret_cast<const float4*>(src)[lane];
  float  s  = v.x + v.y + v.z + v.w;
  float  ss = v.x * v.x + v.y * v.y + v.z * v.z + v.w * v.w;
  s  = wave_sum32(s);
  ss = wave_sum32(ss);
  float mu  = s * (1.f / 128.f);
  float var = ss * (1.f / 128.f) - mu * mu;
  float inv = rsqrtf(var + 1e-5f);
  float4 gv = reinterpret_cast<const float4*>(g)[lane];
  float4 bv = reinterpret_cast<const float4*>(b)[lane];
  union { bf16_t h[4]; uint2 u; } o;
  o.h[0] = (bf16_t)((v.x - mu) * inv * gv.x + bv.x);
  o.h[1] = (bf16_t)((v.y - mu) * inv * gv.y + bv.y);
  o.h[2] = (bf16_t)((v.z - mu) * inv * gv.z + bv.z);
  o.h[3] = (bf16_t)((v.w - mu) * inv * gv.w + bv.w);
  reinterpret_cast<uint2*>(dst)[lane] = o.u;
}

// region id of rolled coordinate (shift-window mask): boundaries D-ws/D-ss etc.
__device__ __forceinline__ int region_of(int gd, int gh, int gw) {
  int rd = (gd < 6)  ? 0 : ((gd < 7)  ? 1 : 2);
  int rh = (gh < 49) ? 0 : ((gh < 53) ? 1 : 2);
  int rw = (gw < 49) ? 0 : ((gw < 53) ? 1 : 2);
  return rd * 9 + rh * 3 + rw;
}

// --------------------------- kernels ---------------------------------------

__global__ void k_cvt_bf16(const float* __restrict__ src, bf16_t* __restrict__ dst, int n) {
  int i = blockIdx.x * 256 + threadIdx.x;
  if (i < n) dst[i] = (bf16_t)src[i];
}

// LN1 + cyclic shift (-1,-3,-3) + window partition + pad to 128 tokens, bf16.
__global__ void __launch_bounds__(128)
k_ln1_shift_part(const float* __restrict__ x, const float* __restrict__ g,
                 const float* __restrict__ b, bf16_t* __restrict__ Xw) {
  int wid  = blockIdx.x * 4 + (threadIdx.x >> 5);   // window-row id
  int lane = threadIdx.x & 31;
  int widx = wid >> 7, t = wid & 127;
  bf16_t* dst = Xw + (size_t)wid * CC;
  if (t >= NTOK) {
    uint2 z = {0u, 0u};
    reinterpret_cast<uint2*>(dst)[lane] = z;
    return;
  }
  int bb = widx >> 8, rr = widx & 255;
  int wd = rr >> 6, wh = (rr >> 3) & 7, ww = rr & 7;
  int dt = t / 49, rem = t - dt * 49, ht = rem / 7, wt = rem - ht * 7;
  int od = (wd * 2 + dt + 1) & 7;                   // (g + shift) mod D
  int oh = wh * 7 + ht + 3; if (oh >= HH)  oh -= HH;
  int ow = ww * 7 + wt + 3; if (ow >= WW_) ow -= WW_;
  const float* src = x + ((((size_t)bb * DD + od) * HH + oh) * WW_ + ow) * CC;
  ln_row(src, g, b, dst, lane);
}

// QKV GEMM: [1024*128,128] x [384,128]^T, 1x4 register-blocked.
// Q pre-scaled by hd^-0.5, V stored transposed (channel x token).
__global__ void __launch_bounds__(256)
k_qkv(const bf16_t* __restrict__ Xw, const bf16_t* __restrict__ Wqkv,
      const float* __restrict__ qb, bf16_t* __restrict__ Qb,
      bf16_t* __restrict__ Kb, bf16_t* __restrict__ Vt) {
  int wave = blockIdx.x * 8 + (threadIdx.x >> 5);
  int lane = threadIdx.x & 31;
  int mtile = wave / 6, grp = wave % 6;              // 24 ntiles -> 6 groups of 4
  v8f acc[4] = {zero8(), zero8(), zero8(), zero8()};
  wmma_rowblk<4>(Xw, CC, Wqkv, CC, mtile * 16, grp * 64, CC, lane, acc);
  int rb = mtile * 16 + ((lane & 16) ? 8 : 0);
#pragma unroll
  for (int t = 0; t < 4; ++t) {
    int   n    = grp * 64 + t * 16 + (lane & 15);
    float bias = qb[n];
    for (int r = 0; r < 8; ++r) {
      int row = rb + r, widx = row >> 7, tok = row & 127;
      float v = acc[t][r] + bias;
      if (n < 128) {
        int head = n >> 5, ch = n & 31;
        Qb[(((size_t)widx * HEADS_ + head) * NPAD + tok) * HD + ch] =
            (bf16_t)(v * 0.17677669529663687f);      // * hd^-0.5
      } else if (n < 256) {
        int nn = n - 128, head = nn >> 5, ch = nn & 31;
        Kb[(((size_t)widx * HEADS_ + head) * NPAD + tok) * HD + ch] = (bf16_t)v;
      } else {
        int nn = n - 256, head = nn >> 5, ch = nn & 31;
        Vt[(((size_t)widx * HEADS_ + head) * HD + ch) * NPAD + tok] = (bf16_t)v;
      }
    }
  }
}

// Attention: block = (window, head), 8 waves, wave s owns rows [16s,16s+16).
// K (8KB) and V (8KB) staged in LDS via async copies; P staged in LDS for
// the operand-layout transpose between softmax and P@V.
__global__ void __launch_bounds__(256)
k_attn(const bf16_t* __restrict__ Qb, const bf16_t* __restrict__ Kb,
       const bf16_t* __restrict__ Vt, const float* __restrict__ rpb,
       bf16_t* __restrict__ Aout) {
  __shared__ __align__(16) bf16_t Ks[NPAD][HD];     // token-major (as Kb)
  __shared__ __align__(16) bf16_t Vs[HD][NPAD];     // channel-major (as Vt)
  __shared__ __align__(16) bf16_t Ps[8][16][NPAD];  // softmaxed probs
  int widx = blockIdx.x >> 2, head = blockIdx.x & 3;
  int s = threadIdx.x >> 5, lane = threadIdx.x & 31;
  const bf16_t* Q = Qb + ((size_t)widx * HEADS_ + head) * NPAD * HD;
  const bf16_t* K = Kb + ((size_t)widx * HEADS_ + head) * NPAD * HD;
  const bf16_t* V = Vt + ((size_t)widx * HEADS_ + head) * HD * NPAD;

  // ---- async stage K,V -> LDS (4 x B128 per thread, offset hits both sides)
  {
    int tid = threadIdx.x;
    unsigned lk = (unsigned)(size_t)(&Ks[0][0]) + (unsigned)tid * 16u;
    unsigned lv = (unsigned)(size_t)(&Vs[0][0]) + (unsigned)tid * 16u;
    unsigned long long gk = (unsigned long long)(const void*)(K + tid * 8);
    unsigned long long gv = (unsigned long long)(const void*)(V + tid * 8);
    asm volatile(
        "global_load_async_to_lds_b128 %0, %2, off\n\t"
        "global_load_async_to_lds_b128 %0, %2, off offset:4096\n\t"
        "global_load_async_to_lds_b128 %1, %3, off\n\t"
        "global_load_async_to_lds_b128 %1, %3, off offset:4096"
        :: "v"(lk), "v"(lv), "v"(gk), "v"(gv)
        : "memory");
  }

  // ---- overlap: analytic relative-position bias + shift-mask setup
  int rr = widx & 255;
  int wd = rr >> 6, wh = (rr >> 3) & 7, ww = rr & 7;
  int rb = (lane & 16) ? 8 : 0;
  int di[8], hi[8], wi[8], regi[8];
  for (int r = 0; r < 8; ++r) {
    int i = s * 16 + rb + r;
    int d = i / 49, rem = i - d * 49, h = rem / 7, w = rem - h * 7;
    di[r] = d; hi[r] = h; wi[r] = w;
    regi[r] = region_of(wd * 2 + d, wh * 7 + h, ww * 7 + w);
  }

  // Q fragment: one A-operand load serves all 8 S tiles of this strip.
  v16bf aq = ld_a16(Q + (size_t)(s * 16 + (lane & 15)) * HD + ((lane & 16) ? 8 : 0));

  asm volatile("s_wait_asynccnt 0x0" ::: "memory");
  __syncthreads();

  // ---- S = Q K^T from LDS (K-depth 32 = one WMMA per 16x16 tile)
  v8f acc[8];
#pragma unroll
  for (int t = 0; t < 8; ++t) {
    v16bf b = ld_b16(&Ks[t * 16 + (lane & 15)][(lane & 16) ? 8 : 0]);
    acc[t] = __builtin_amdgcn_wmma_f32_16x16x32_bf16(false, aq, false, b, (short)0,
                                                     zero8(), false, false);
  }

  // ---- bias + mask
  for (int t = 0; t < 8; ++t) {
    int j = t * 16 + (lane & 15);
    int d = j / 49, rem = j - d * 49, h = rem / 7, w = rem - h * 7;
    int regj = region_of(wd * 2 + d, wh * 7 + h, ww * 7 + w);
    bool jpad = (j >= NTOK);
    for (int r = 0; r < 8; ++r) {
      if (jpad) {
        acc[t][r] = -1e30f;                         // padded key: excluded
      } else {
        int idx = (di[r] - d + 1) * 169 + (hi[r] - h + 6) * 13 + (wi[r] - w + 6);
        float bias = rpb[idx * HEADS_ + head];
        acc[t][r] += bias + ((regi[r] != regj) ? -100.f : 0.f);
      }
    }
  }

  // ---- softmax along 128 keys (8 col-tiles x 16 lanes per half-wave)
  for (int r = 0; r < 8; ++r) {
    float m = -3.0e38f;
    for (int t = 0; t < 8; ++t) m = fmaxf(m, acc[t][r]);
    for (int msk = 1; msk < 16; msk <<= 1) m = fmaxf(m, __shfl_xor(m, msk, 32));
    float sum = 0.f;
    for (int t = 0; t < 8; ++t) { float e = __expf(acc[t][r] - m); acc[t][r] = e; sum += e; }
    for (int msk = 1; msk < 16; msk <<= 1) sum += __shfl_xor(sum, msk, 32);
    float inv = 1.f / sum;
    for (int t = 0; t < 8; ++t) acc[t][r] *= inv;
  }

  // ---- transpose P (C/D layout) -> LDS row-major, re-read in A layout
  for (int t = 0; t < 8; ++t)
    for (int r = 0; r < 8; ++r)
      Ps[s][rb + r][t * 16 + (lane & 15)] = (bf16_t)acc[t][r];
  __syncthreads();

  // P fragments hoisted across both output channel tiles.
  v16bf pa[4];
#pragma unroll
  for (int kt = 0; kt < 4; ++kt)
    pa[kt] = ld_a16(&Ps[s][lane & 15][kt * 32 + ((lane & 16) ? 8 : 0)]);

  // ---- out = P @ V : K = 128 keys (padded keys carry P = 0)
#pragma unroll
  for (int nt = 0; nt < 2; ++nt) {
    v8f o = zero8();
#pragma unroll
    for (int kt = 0; kt < 4; ++kt) {
      v16bf b = ld_b16(&Vs[nt * 16 + (lane & 15)][kt * 32 + ((lane & 16) ? 8 : 0)]);
      o = __builtin_amdgcn_wmma_f32_16x16x32_bf16(false, pa[kt], false, b, (short)0,
                                                  o, false, false);
    }
    int ch = head * HD + nt * 16 + (lane & 15);
    for (int r = 0; r < 8; ++r) {
      int tok = s * 16 + rb + r;
      Aout[((size_t)widx * NPAD + tok) * CC + ch] = (bf16_t)o[r];
    }
  }
}

// proj GEMM + window-reverse + un-shift + residual: d_out = x + proj(attn)
__global__ void __launch_bounds__(256)
k_proj(const bf16_t* __restrict__ Aout, const bf16_t* __restrict__ Wp,
       const float* __restrict__ pb, const float* __restrict__ xin,
       float* __restrict__ out) {
  int wave = blockIdx.x * 8 + (threadIdx.x >> 5);
  int lane = threadIdx.x & 31;
  int mtile = wave >> 1, grp = wave & 1;             // 8 ntiles -> 2 groups of 4
  v8f acc[4] = {zero8(), zero8(), zero8(), zero8()};
  wmma_rowblk<4>(Aout, CC, Wp, CC, mtile * 16, grp * 64, CC, lane, acc);
  int rbg = mtile * 16 + ((lane & 16) ? 8 : 0);
#pragma unroll
  for (int t = 0; t < 4; ++t) {
    int   n    = grp * 64 + t * 16 + (lane & 15);
    float bias = pb[n];
    for (int r = 0; r < 8; ++r) {
      int row = rbg + r, widx = row >> 7, tok = row & 127;
      if (tok >= NTOK) continue;                     // padded rows dropped
      int bb = widx >> 8, rr = widx & 255;
      int wd = rr >> 6, wh = (rr >> 3) & 7, ww = rr & 7;
      int dt = tok / 49, rem = tok - dt * 49, ht = rem / 7, wt = rem - ht * 7;
      int od = (wd * 2 + dt + 1) & 7;
      int oh = wh * 7 + ht + 3; if (oh >= HH)  oh -= HH;
      int ow = ww * 7 + wt + 3; if (ow >= WW_) ow -= WW_;
      size_t gi = ((((size_t)bb * DD + od) * HH + oh) * WW_ + ow) * CC + n;
      out[gi] = xin[gi] + acc[t][r] + bias;
    }
  }
}

__global__ void __launch_bounds__(128)
k_ln2(const float* __restrict__ xr, const float* __restrict__ g,
      const float* __restrict__ b, bf16_t* __restrict__ dst) {
  int row  = blockIdx.x * 4 + (threadIdx.x >> 5);
  int lane = threadIdx.x & 31;
  ln_row(xr + (size_t)row * CC, g, b, dst + (size_t)row * CC, lane);
}

// fc1 + exact GELU, one MLP chunk (hidden chunk stays L2-resident)
__global__ void __launch_bounds__(256)
k_fc1(const bf16_t* __restrict__ Achunk, const bf16_t* __restrict__ W1,
      const float* __restrict__ b1, bf16_t* __restrict__ Hid) {
  int wave = blockIdx.x * 8 + (threadIdx.x >> 5);
  int lane = threadIdx.x & 31;
  int mt = wave >> 3, grp = wave & 7;                // 32 ntiles -> 8 groups
  v8f acc[4] = {zero8(), zero8(), zero8(), zero8()};
  wmma_rowblk<4>(Achunk, CC, W1, CC, mt * 16, grp * 64, CC, lane, acc);
  int rb = mt * 16 + ((lane & 16) ? 8 : 0);
#pragma unroll
  for (int t = 0; t < 4; ++t) {
    int   n    = grp * 64 + t * 16 + (lane & 15);
    float bias = b1[n];
    for (int r = 0; r < 8; ++r) {
      float v = acc[t][r] + bias;
      float gel = 0.5f * v * (1.f + erff(v * 0.70710678118654752f));
      Hid[(size_t)(rb + r) * HIDDEN + n] = (bf16_t)gel;
    }
  }
}

// fc2 + residual add into d_out
__global__ void __launch_bounds__(256)
k_fc2(const bf16_t* __restrict__ Hid, const bf16_t* __restrict__ W2,
      const float* __restrict__ b2, float* __restrict__ out, int rowOff) {
  int wave = blockIdx.x * 8 + (threadIdx.x >> 5);
  int lane = threadIdx.x & 31;
  int mt = wave >> 1, grp = wave & 1;                // 8 ntiles -> 2 groups
  v8f acc[4] = {zero8(), zero8(), zero8(), zero8()};
  wmma_rowblk<4>(Hid, HIDDEN, W2, HIDDEN, mt * 16, grp * 64, HIDDEN, lane, acc);
  int rb = mt * 16 + ((lane & 16) ? 8 : 0);
#pragma unroll
  for (int t = 0; t < 4; ++t) {
    int   n    = grp * 64 + t * 16 + (lane & 15);
    float bias = b2[n];
    for (int r = 0; r < 8; ++r) {
      size_t gi = (size_t)(rowOff + rb + r) * CC + n;
      out[gi] += acc[t][r] + bias;
    }
  }
}

// --------------------------- launcher --------------------------------------

extern "C" void kernel_launch(void* const* d_in, const int* in_sizes, int n_in,
                              void* d_out, int out_size, void* d_ws, size_t ws_size,
                              hipStream_t stream) {
  (void)in_sizes; (void)n_in; (void)out_size; (void)ws_size;
  const float* x      = (const float*)d_in[0];
  const float* n1w    = (const float*)d_in[1];
  const float* n1b    = (const float*)d_in[2];
  const float* qkv_w  = (const float*)d_in[3];
  const float* qkv_b  = (const float*)d_in[4];
  const float* rpb    = (const float*)d_in[5];
  const float* proj_w = (const float*)d_in[6];
  const float* proj_b = (const float*)d_in[7];
  const float* n2w    = (const float*)d_in[8];
  const float* n2b    = (const float*)d_in[9];
  const float* fc1_w  = (const float*)d_in[10];
  const float* fc1_b  = (const float*)d_in[11];
  const float* fc2_w  = (const float*)d_in[12];
  const float* fc2_b  = (const float*)d_in[13];
  float* out = (float*)d_out;

  // workspace layout (bf16 elements); Aout reuses Xw, LN2 output reuses Qb.
  bf16_t* wsb   = (bf16_t*)d_ws;
  bf16_t* Wqkv  = wsb;                       // 384*128
  bf16_t* Wproj = Wqkv  + 384 * 128;         // 128*128
  bf16_t* Wfc1  = Wproj + 128 * 128;         // 512*128
  bf16_t* Wfc2  = Wfc1  + 512 * 128;         // 128*512
  bf16_t* Xw    = Wfc2  + 128 * 512;         // 1024*128*128
  bf16_t* Qb    = Xw + (size_t)NWIN * NPAD * CC;
  bf16_t* Kb    = Qb + (size_t)NWIN * NPAD * CC;
  bf16_t* Vt    = Kb + (size_t)NWIN * NPAD * CC;
  bf16_t* Hid   = Vt + (size_t)NWIN * NPAD * CC;   // 12544*512 chunk
  bf16_t* Aout  = Xw;                              // reuse (Xw consumed by qkv)
  bf16_t* Ln2   = Qb;                              // reuse (Qb consumed by attn)

  // weights -> bf16 (tiny, L2-resident)
  k_cvt_bf16<<<(384 * 128 + 255) / 256, 256, 0, stream>>>(qkv_w, Wqkv, 384 * 128);
  k_cvt_bf16<<<(128 * 128 + 255) / 256, 256, 0, stream>>>(proj_w, Wproj, 128 * 128);
  k_cvt_bf16<<<(512 * 128 + 255) / 256, 256, 0, stream>>>(fc1_w, Wfc1, 512 * 128);
  k_cvt_bf16<<<(128 * 512 + 255) / 256, 256, 0, stream>>>(fc2_w, Wfc2, 128 * 512);

  // LN1 + shift + partition: 1024*128 rows, 4 waves/block
  k_ln1_shift_part<<<NWIN * NPAD / 4, 128, 0, stream>>>(x, n1w, n1b, Xw);

  // QKV: 8192 mtiles x 6 n-groups, 8 wave-blocks per CTA
  k_qkv<<<(NWIN * NPAD / 16) * 6 / 8, 256, 0, stream>>>(Xw, Wqkv, qkv_b, Qb, Kb, Vt);

  // attention: one block per (window, head)
  k_attn<<<NWIN * HEADS_, 256, 0, stream>>>(Qb, Kb, Vt, rpb, Aout);

  // proj + reverse + residual -> d_out holds x1
  k_proj<<<(NWIN * NPAD / 16) * 2 / 8, 256, 0, stream>>>(Aout, Wproj, proj_b, x, out);

  // LN2 over all tokens
  k_ln2<<<TOKENS / 4, 128, 0, stream>>>(out, n2w, n2b, Ln2);

  // MLP, chunked so the bf16 hidden (12.8MB) stays in the 192MB L2
  for (int c = 0; c < 8; ++c) {
    int rowOff = c * MLP_CHUNK_ROWS;
    k_fc1<<<(MLP_CHUNK_ROWS / 16) * 8 / 8, 256, 0, stream>>>(
        Ln2 + (size_t)rowOff * CC, Wfc1, fc1_b, Hid);
    k_fc2<<<(MLP_CHUNK_ROWS / 16) * 2 / 8, 256, 0, stream>>>(
        Hid, Wfc2, fc2_b, out, rowOff);
  }
}
